// VQVAE_IDX_85950885528547
// MI455X (gfx1250) — matrix operand order repaired and checked by reading, not compiled, not run on hardware
//
#include <hip/hip_runtime.h>

// ---------------------------------------------------------------------------
// CDNA5 (gfx1250) VQ-VAE forward.
// Wave32 WMMA bf16 16x16x32 (f32 accum); double-buffered async global->LDS
// staging (ASYNCcnt software pipeline); operands pre-converted to bf16
// (weights transposed to N-major).
// ---------------------------------------------------------------------------

typedef __attribute__((ext_vector_type(16))) __bf16          v16bf;
typedef __attribute__((ext_vector_type(16))) unsigned short  v16s;
typedef __attribute__((ext_vector_type(8)))  float           v8f;

#define BATCH   8192
#define DIN     1024
#define DE      2048
#define NRES    4
#define NE      2048
#define PP      4
#define ED      512
#define HIDDIM  128
#define ODIM    64

// fp32 -> bf16 round-to-nearest-even (bit trick)
__device__ __forceinline__ unsigned short f2bf(float f) {
    unsigned u = __builtin_bit_cast(unsigned, f);
    u += 0x7fffu + ((u >> 16) & 1u);
    return (unsigned short)(u >> 16);
}

// orderable-uint encoding of float (ascending float -> ascending uint)
__device__ __forceinline__ unsigned ford(float f) {
    unsigned u = __builtin_bit_cast(unsigned, f);
    return (u & 0x80000000u) ? ~u : (u | 0x80000000u);
}

// 16B async copy global -> LDS (ASYNCcnt tracked, bypasses VGPRs).
__device__ __forceinline__ void async_b128(unsigned lds_addr, const void* gaddr) {
    asm volatile("global_load_async_to_lds_b128 %0, %1, off"
                 :: "v"(lds_addr), "v"(gaddr) : "memory");
}
// wait until at most N async ops outstanding (in-order retirement)
template <int N>
__device__ __forceinline__ void wait_async() {
    asm volatile("s_wait_asynccnt %0" :: "i"(N) : "memory");
}

// A-fragment (16-bit A 16x32): lane half kb in {0,16};
// per lane 16 bf16: K = kb..kb+15 of its row.
__device__ __forceinline__ v16bf frag_a(const unsigned short* rowbase, int kb) {
    v16s t;
#pragma unroll
    for (int i = 0; i < 8; ++i) {
        t[i]     = rowbase[kb + i];
        t[i + 8] = rowbase[kb + 8 + i];
    }
    return __builtin_bit_cast(v16bf, t);
}

// B-fragment (B 32x16): lanes 0-15 hold K=0..15, lanes 16-31 K=16..31 of
// column N=lane&15; B stored in LDS transposed [n][k] -> contiguous 32B.
__device__ __forceinline__ v16bf frag_b(const unsigned short* rowbase, int kh) {
    v16s t;
#pragma unroll
    for (int i = 0; i < 16; ++i) t[i] = rowbase[kh * 16 + i];
    return __builtin_bit_cast(v16bf, t);
}

// ---------------------------------------------------------------------------
// WMMA GEMM: C[M,N] = [A1|A2] @ Bt^T (+ bias)
//   A1/A2: bf16 row-major (split at column K1; K1 % 32 == 0, A2 may be null).
//   Bt:    bf16 (N x K) row-major (weights pre-transposed).
//   EPI==0: C = .. + bias (f32 store)
//   EPI==1: as 0, plus bf16 mirror store to Cbf
//   EPI==2: VQ argmin: dist = cbnorm[col] - 2*S; atomicMin u64 (dist,idx) key.
// Block: 256 thr = 8 waves (4 M-waves x 2 N-waves), tile 128x128, K-step 32.
// Double-buffered LDS; next tile's asyncs overlap current tile's WMMAs.
// ---------------------------------------------------------------------------
template <int EPI>
__global__ __launch_bounds__(256) void k_gemm(
    const unsigned short* __restrict__ A1, const unsigned short* __restrict__ A2,
    int K1, int lda1, int lda2,
    const unsigned short* __restrict__ Bt, int ldb,
    const float* __restrict__ bias,
    float* __restrict__ C, unsigned short* __restrict__ Cbf, int ldc,
    int M, int N, int K,
    const float* __restrict__ cbnorm, unsigned long long* __restrict__ keys)
{
    __shared__ unsigned short As[2][128][40];   // pad -> 80B rows (16B aligned)
    __shared__ unsigned short Bs[2][128][40];   // B transposed: [n][k]

    const int tid   = threadIdx.x;
    const int lane  = tid & 31;                 // wave32
    const int wid   = tid >> 5;
    const int waveM = wid & 3;
    const int waveN = wid >> 2;
    const int bm    = blockIdx.y * 128;
    const int bn    = blockIdx.x * 128;

    const int srow = tid >> 1;                  // staging row (A) / n (B)
    const int c0   = (tid & 1) * 16;            // staging K-chunk: 0 or 16

    // stage K-tile kt into LDS buffer buf (4 async b128 per thread)
    auto stage = [&](int buf, int kt) {
        const int c = kt + c0;
        const unsigned short* asrc =
            (c < K1) ? (A1 + (size_t)(bm + srow) * lda1 + c)
                     : (A2 + (size_t)(bm + srow) * lda2 + (c - K1));
        const unsigned short* bsrc = Bt + (size_t)(bn + srow) * ldb + c;
        if (kt + 64 < K) {                      // global_prefetch_b8 tile kt+64
            __builtin_prefetch(asrc + 64, 0, 3);
            __builtin_prefetch(bsrc + 64, 0, 3);
        }
        const unsigned aoff = (unsigned)(size_t)&As[buf][srow][c0];
        const unsigned boff = (unsigned)(size_t)&Bs[buf][srow][c0];
        async_b128(aoff,      asrc);
        async_b128(aoff + 16, asrc + 8);
        async_b128(boff,      bsrc);
        async_b128(boff + 16, bsrc + 8);
    };

    v8f acc[2][4] = {};

    stage(0, 0);                                // prologue: tile 0 -> buf 0
    for (int kt = 0; kt < K; kt += 32) {
        const int cur = (kt >> 5) & 1;
        const bool more = (kt + 32) < K;
        if (more) stage(cur ^ 1, kt + 32);      // overlaps with compute below
        if (more) wait_async<4>();              // cur's 4 asyncs done; 4 in flight
        else      wait_async<0>();
        __syncthreads();                        // all waves' cur staging visible

        const int mrow = waveM * 32 + (lane & 15);
        const int nrow = waveN * 64 + (lane & 15);
        const int kb   = (lane >> 4) * 16;
        const int kh   = (lane >> 4);
        v16bf af[2], bfr[4];
        af[0] = frag_a(&As[cur][mrow][0], kb);
        af[1] = frag_a(&As[cur][mrow + 16][0], kb);
#pragma unroll
        for (int ni = 0; ni < 4; ++ni)
            bfr[ni] = frag_b(&Bs[cur][nrow + ni * 16][0], kh);
#pragma unroll
        for (int mi = 0; mi < 2; ++mi)
#pragma unroll
            for (int ni = 0; ni < 4; ++ni)
                acc[mi][ni] = __builtin_amdgcn_wmma_f32_16x16x32_bf16(
                    false, af[mi], false, bfr[ni], (short)0, acc[mi][ni],
                    false, false);
        __syncthreads();                        // cur reusable for staging kt+64
    }

    // ---- epilogue (C layout: VGPR r -> M=r lanes 0-15, M=r+8 lanes 16-31) ----
    const int col0 = bn + waveN * 64 + (lane & 15);
    const int row0 = bm + waveM * 32 + ((lane >> 4) * 8);
    if (EPI <= 1) {
        float bvec[4];
#pragma unroll
        for (int ni = 0; ni < 4; ++ni) bvec[ni] = bias[col0 + ni * 16];
#pragma unroll
        for (int mi = 0; mi < 2; ++mi)
#pragma unroll
            for (int r = 0; r < 8; ++r) {
                const int row = row0 + mi * 16 + r;
                float* crow = C + (size_t)row * ldc;
                unsigned short* brow =
                    (EPI == 1) ? (Cbf + (size_t)row * ldc) : nullptr;
#pragma unroll
                for (int ni = 0; ni < 4; ++ni) {
                    const int col = col0 + ni * 16;
                    const float v = acc[mi][ni][r] + bvec[ni];
                    crow[col] = v;
                    if (EPI == 1) brow[col] = f2bf(v);
                }
            }
    } else {
        float nvec[4];
#pragma unroll
        for (int ni = 0; ni < 4; ++ni) nvec[ni] = cbnorm[col0 + ni * 16];
#pragma unroll
        for (int mi = 0; mi < 2; ++mi)
#pragma unroll
            for (int r = 0; r < 8; ++r) {
                const int row = row0 + mi * 16 + r;
                float best = 3.4e38f;
                int bidx = 0;
#pragma unroll
                for (int ni = 0; ni < 4; ++ni) {
                    const float d = nvec[ni] - 2.0f * acc[mi][ni][r];
                    if (d < best) { best = d; bidx = col0 + ni * 16; }
                }
                const unsigned long long key =
                    ((unsigned long long)ford(best) << 32) | (unsigned)bidx;
                atomicMin(&keys[row], key);
            }
    }
}

// ---------------------------------------------------------------------------
// h = h + relu(LayerNorm(t)*g + b); also maintains bf16 mirror of h.
// ---------------------------------------------------------------------------
__global__ __launch_bounds__(256) void k_ln_relu_add(
    const float* __restrict__ t, float* __restrict__ h,
    unsigned short* __restrict__ hbf,
    const float* __restrict__ g, const float* __restrict__ be, int n)
{
    __shared__ float s1[256], s2[256];
    const int row = blockIdx.x;
    const float* tr = t + (size_t)row * n;
    float a = 0.f, b = 0.f;
    for (int c = threadIdx.x; c < n; c += 256) {
        const float v = tr[c];
        a += v; b += v * v;
    }
    s1[threadIdx.x] = a; s2[threadIdx.x] = b;
    __syncthreads();
    for (int off = 128; off > 0; off >>= 1) {
        if (threadIdx.x < off) {
            s1[threadIdx.x] += s1[threadIdx.x + off];
            s2[threadIdx.x] += s2[threadIdx.x + off];
        }
        __syncthreads();
    }
    const float mean = s1[0] / (float)n;
    const float var  = s2[0] / (float)n - mean * mean;
    const float inv  = rsqrtf(var + 1e-5f);
    float* hr = h + (size_t)row * n;
    unsigned short* hb = hbf + (size_t)row * n;
    for (int c = threadIdx.x; c < n; c += 256) {
        const float y  = (tr[c] - mean) * inv * g[c] + be[c];
        const float hn = hr[c] + fmaxf(y, 0.f);
        hr[c] = hn;
        hb[c] = f2bf(hn);
    }
}

// ---------------------------------------------------------------------------
// IdxEmbedding tiny MLP -> bf16 output (consumed as GEMM A2)
// ---------------------------------------------------------------------------
__global__ __launch_bounds__(128) void k_idx_emb(
    const int* __restrict__ bi,
    const float* __restrict__ le, const float* __restrict__ te,
    const float* __restrict__ we,
    const float* __restrict__ fcw, const float* __restrict__ fcb,
    const float* __restrict__ ow,  const float* __restrict__ ob,
    unsigned short* __restrict__ out)
{
    __shared__ float hid[HIDDIM];
    const int row = blockIdx.x;
    const int j   = threadIdx.x;
    const int c0 = bi[row * 6 + 0], c1 = bi[row * 6 + 1], c2 = bi[row * 6 + 2];
    float acc = fcb[j];
    for (int k = 0; k < HIDDIM; ++k) acc += le[c0 * HIDDIM + k] * fcw[k * HIDDIM + j];
    for (int k = 0; k < HIDDIM; ++k) acc += te[c1 * HIDDIM + k] * fcw[(HIDDIM + k) * HIDDIM + j];
    for (int k = 0; k < HIDDIM; ++k) acc += we[c2 * HIDDIM + k] * fcw[(2 * HIDDIM + k) * HIDDIM + j];
    for (int k = 0; k < 3; ++k)
        acc += (float)bi[row * 6 + 3 + k] * fcw[(3 * HIDDIM + k) * HIDDIM + j];
    hid[j] = acc;
    __syncthreads();
    if (j < ODIM) {
        float o = ob[j];
        for (int k = 0; k < HIDDIM; ++k) o += hid[k] * ow[k * ODIM + j];
        out[(size_t)row * ODIM + j] = f2bf(o);
    }
}

// ---------------------------------------------------------------------------
// one-time prep kernels
// ---------------------------------------------------------------------------
__global__ void k_xprep(const float* __restrict__ x, const float* __restrict__ scale,
                        const float* __restrict__ shift, unsigned short* __restrict__ xbf,
                        int total, int n)
{
    const int i = blockIdx.x * 256 + threadIdx.x;
    if (i < total) {
        const int c = i % n;
        xbf[i] = f2bf((x[i] - shift[c]) / scale[c]);
    }
}

// W (K x N) f32 -> Wt (N x K) bf16
__global__ void k_wt_bf(const float* __restrict__ W, unsigned short* __restrict__ Wt,
                        int K, int N)
{
    const int i = blockIdx.x * 256 + threadIdx.x;
    if (i < K * N) {
        const int k = i / N, n = i % N;
        Wt[(size_t)n * K + k] = f2bf(W[i]);
    }
}

__global__ void k_cb_bf(const float* __restrict__ cb, unsigned short* __restrict__ cbbf,
                        int total)
{
    const int i = blockIdx.x * 256 + threadIdx.x;
    if (i < total) cbbf[i] = f2bf(cb[i]);
}

__global__ void k_cbnorm(const float* __restrict__ cb, float* __restrict__ nrm,
                         int ne, int ed)
{
    const int n = blockIdx.x * 256 + threadIdx.x;
    if (n < ne) {
        float s = 0.f;
        for (int k = 0; k < ed; ++k) {
            const float v = cb[(size_t)n * ed + k];
            s += v * v;
        }
        nrm[n] = s;
    }
}

__global__ void k_init(unsigned long long* keys, float* hist, float* lossacc,
                       int nk, int nh)
{
    const int i = blockIdx.x * 256 + threadIdx.x;
    if (i < nk) keys[i] = ~0ull;
    if (i < nh) hist[i] = 0.f;
    if (i == 0) *lossacc = 0.f;
}

// per zf-row: idx extract, z_q gather (f32 out + bf16 mirror), loss, histogram
__global__ __launch_bounds__(128) void k_vq_fin(
    const unsigned long long* __restrict__ keys, const float* __restrict__ cb,
    const float* __restrict__ ze, float* __restrict__ zq,
    unsigned short* __restrict__ zqbf,
    float* __restrict__ idx_out, float* __restrict__ hist,
    float* __restrict__ lossacc)
{
    __shared__ float s[128];
    const int row = blockIdx.x;
    const int idx = (int)(unsigned)(keys[row] & 0xffffffffull);
    float a = 0.f;
    for (int k = threadIdx.x; k < ED; k += 128) {
        const float q = cb[(size_t)idx * ED + k];
        zq[(size_t)row * ED + k]   = q;
        zqbf[(size_t)row * ED + k] = f2bf(q);
        const float d = q - ze[(size_t)row * ED + k];
        a += d * d;
    }
    s[threadIdx.x] = a;
    __syncthreads();
    for (int off = 64; off > 0; off >>= 1) {
        if (threadIdx.x < off) s[threadIdx.x] += s[threadIdx.x + off];
        __syncthreads();
    }
    if (threadIdx.x == 0) {
        atomicAdd(lossacc, s[0]);
        atomicAdd(&hist[idx], 1.0f);
        idx_out[row] = (float)idx;
    }
}

__global__ __launch_bounds__(256) void k_perp_loss(
    const float* __restrict__ hist, const float* __restrict__ lossacc,
    float* __restrict__ o_loss, float* __restrict__ o_perp)
{
    __shared__ float s[256];
    float a = 0.f;
    for (int i = threadIdx.x; i < NE; i += 256) {
        const float p = hist[i] / (float)(BATCH * PP);
        a += p * logf(p + 1e-10f);
    }
    s[threadIdx.x] = a;
    __syncthreads();
    for (int off = 128; off > 0; off >>= 1) {
        if (threadIdx.x < off) s[threadIdx.x] += s[threadIdx.x + off];
        __syncthreads();
    }
    if (threadIdx.x == 0) {
        *o_perp = expf(-s[0]);
        *o_loss = 1.25f * (*lossacc) / ((float)BATCH * (float)DE);
    }
}

__global__ void k_xhat(const float* __restrict__ t, const float* __restrict__ scale,
                       const float* __restrict__ shift, float* __restrict__ out,
                       int total, int n)
{
    const int i = blockIdx.x * 256 + threadIdx.x;
    if (i < total) {
        const int c = i % n;
        out[i] = scale[c] * t[i] + shift[c];
    }
}

// ---------------------------------------------------------------------------
extern "C" void kernel_launch(void* const* d_in, const int* in_sizes, int n_in,
                              void* d_out, int out_size, void* d_ws, size_t ws_size,
                              hipStream_t stream)
{
    (void)in_sizes; (void)n_in; (void)out_size; (void)ws_size;
    const float* x          = (const float*)d_in[0];
    const int*   bidx       = (const int*)  d_in[1];
    const float* scale      = (const float*)d_in[2];
    const float* shift      = (const float*)d_in[3];
    const float* enc_in_w   = (const float*)d_in[4];
    const float* enc_in_b   = (const float*)d_in[5];
    const float* enc_res_w  = (const float*)d_in[6];
    const float* enc_res_b  = (const float*)d_in[7];
    const float* enc_ln_g   = (const float*)d_in[8];
    const float* enc_ln_b   = (const float*)d_in[9];
    const float* idx_le     = (const float*)d_in[10];
    const float* idx_te     = (const float*)d_in[11];
    const float* idx_we     = (const float*)d_in[12];
    const float* idx_fc_w   = (const float*)d_in[13];
    const float* idx_fc_b   = (const float*)d_in[14];
    const float* idx_out_w  = (const float*)d_in[15];
    const float* idx_out_b  = (const float*)d_in[16];
    const float* idxcat_w   = (const float*)d_in[17];
    const float* idxcat_b   = (const float*)d_in[18];
    const float* codebook   = (const float*)d_in[19];
    const float* dec_in_w   = (const float*)d_in[20];
    const float* dec_in_b   = (const float*)d_in[21];
    const float* dec_res_w  = (const float*)d_in[22];
    const float* dec_res_b  = (const float*)d_in[23];
    const float* dec_ln_g   = (const float*)d_in[24];
    const float* dec_ln_b   = (const float*)d_in[25];
    const float* dec_out_w  = (const float*)d_in[26];
    const float* dec_out_b  = (const float*)d_in[27];

    // ---- workspace layout (float units, 32-float aligned blocks) ----
    float* ws = (float*)d_ws;
    size_t o = 0;
    auto allocf = [&](size_t nfloats) -> float* {
        float* p = ws + o;
        o += (nfloats + 31) & ~(size_t)31;
        return p;
    };
    float* f_h     = allocf((size_t)BATCH * DE);      // hidden f32
    float* f_t     = allocf((size_t)BATCH * DE);      // GEMM temp f32
    float* f_ze    = allocf((size_t)BATCH * DE);      // z_e f32 (for loss)
    unsigned long long* keys = (unsigned long long*)allocf((size_t)BATCH * PP * 2);
    float* hist    = allocf(NE);
    float* lossacc = allocf(32);
    float* cbn     = allocf(NE);
    auto allocbf = [&](size_t nelems) -> unsigned short* {
        return (unsigned short*)allocf((nelems + 1) / 2);
    };
    unsigned short* x_bf   = allocbf((size_t)BATCH * DIN);
    unsigned short* h_bf   = allocbf((size_t)BATCH * DE);
    unsigned short* ze_bf  = allocbf((size_t)BATCH * DE);
    unsigned short* zq_bf  = allocbf((size_t)BATCH * DE);
    unsigned short* ix_bf  = allocbf((size_t)BATCH * ODIM);
    unsigned short* w_ein  = allocbf((size_t)DE * DIN);        // enc_in_w^T
    unsigned short* w_eres = allocbf((size_t)DE * DE);         // enc_res_w^T
    unsigned short* w_cat  = allocbf((size_t)DE * (DE + ODIM));// idxcat_w^T
    unsigned short* w_cb   = allocbf((size_t)NE * ED);         // codebook bf16
    unsigned short* w_din  = allocbf((size_t)DE * DE);         // dec_in_w^T
    unsigned short* w_dres = allocbf((size_t)DE * DE);         // dec_res_w^T
    unsigned short* w_dout = allocbf((size_t)DIN * DE);        // dec_out_w^T

    // ---- output layout (tuple flattened, float) ----
    float* out    = (float*)d_out;
    float* o_loss = out;
    float* o_xhat = out + 1;
    float* o_perp = out + 1 + (size_t)BATCH * DIN;
    float* o_zq   = o_perp + 1;
    float* o_idx  = o_zq + (size_t)BATCH * DE;

    const dim3 blk(256);
    const dim3 gEnc(DE / 128, BATCH / 128);       // 16 x 64
    const dim3 gOut(DIN / 128, BATCH / 128);      // 8 x 64
    const dim3 gVq(NE / 128, (BATCH * PP) / 128); // 16 x 256
    auto cdiv = [](size_t a, size_t b) { return (unsigned)((a + b - 1) / b); };

    // ---- one-time bf16 prep ----
    k_xprep<<<cdiv((size_t)BATCH * DIN, 256), blk, 0, stream>>>(
        x, scale, shift, x_bf, BATCH * DIN, DIN);
    k_wt_bf<<<cdiv((size_t)DIN * DE, 256), blk, 0, stream>>>(enc_in_w, w_ein, DIN, DE);
    k_wt_bf<<<cdiv((size_t)DE * DE, 256), blk, 0, stream>>>(enc_res_w, w_eres, DE, DE);
    k_wt_bf<<<cdiv((size_t)(DE + ODIM) * DE, 256), blk, 0, stream>>>(
        idxcat_w, w_cat, DE + ODIM, PP * ED);
    k_cb_bf<<<cdiv((size_t)NE * ED, 256), blk, 0, stream>>>(codebook, w_cb, NE * ED);
    k_wt_bf<<<cdiv((size_t)DE * DE, 256), blk, 0, stream>>>(dec_in_w, w_din, DE, DE);
    k_wt_bf<<<cdiv((size_t)DE * DE, 256), blk, 0, stream>>>(dec_res_w, w_dres, DE, DE);
    k_wt_bf<<<cdiv((size_t)DE * DIN, 256), blk, 0, stream>>>(dec_out_w, w_dout, DE, DIN);
    k_cbnorm<<<cdiv(NE, 256), blk, 0, stream>>>(codebook, cbn, NE, ED);
    k_init<<<cdiv(BATCH * PP, 256), blk, 0, stream>>>(keys, hist, lossacc,
                                                      BATCH * PP, NE);

    // ---- encoder ----
    k_gemm<1><<<gEnc, blk, 0, stream>>>(x_bf, nullptr, DIN, DIN, 0,
                                        w_ein, DIN, enc_in_b,
                                        f_h, h_bf, DE,
                                        BATCH, DE, DIN, nullptr, nullptr);
    for (int r = 0; r < NRES; ++r) {
        k_gemm<0><<<gEnc, blk, 0, stream>>>(h_bf, nullptr, DE, DE, 0,
                                            w_eres, DE, enc_res_b,
                                            f_t, nullptr, DE,
                                            BATCH, DE, DE, nullptr, nullptr);
        k_ln_relu_add<<<BATCH, blk, 0, stream>>>(f_t, f_h, h_bf,
                                                 enc_ln_g, enc_ln_b, DE);
    }

    // ---- IdxEmbedding ----
    k_idx_emb<<<BATCH, 128, 0, stream>>>(bidx, idx_le, idx_te, idx_we,
                                         idx_fc_w, idx_fc_b, idx_out_w, idx_out_b,
                                         ix_bf);

    // ---- z_e = [h | out_idx] @ idxcat_w + b ----
    k_gemm<1><<<gEnc, blk, 0, stream>>>(h_bf, ix_bf, DE, DE, ODIM,
                                        w_cat, DE + ODIM, idxcat_b,
                                        f_ze, ze_bf, PP * ED,
                                        BATCH, PP * ED, DE + ODIM, nullptr, nullptr);

    // ---- vector quantizer ----
    k_gemm<2><<<gVq, blk, 0, stream>>>(ze_bf, nullptr, ED, ED, 0,
                                       w_cb, ED, nullptr,
                                       nullptr, nullptr, 0,
                                       BATCH * PP, NE, ED, cbn, keys);
    k_vq_fin<<<BATCH * PP, 128, 0, stream>>>(keys, codebook, f_ze, o_zq, zq_bf,
                                             o_idx, hist, lossacc);
    k_perp_loss<<<1, blk, 0, stream>>>(hist, lossacc, o_loss, o_perp);

    // ---- decoder (z_q_st == z_q numerically) ----
    k_gemm<1><<<gEnc, blk, 0, stream>>>(zq_bf, nullptr, PP * ED, PP * ED, 0,
                                        w_din, PP * ED, dec_in_b,
                                        f_h, h_bf, DE,
                                        BATCH, DE, PP * ED, nullptr, nullptr);
    for (int r = 0; r < NRES; ++r) {
        k_gemm<0><<<gEnc, blk, 0, stream>>>(h_bf, nullptr, DE, DE, 0,
                                            w_dres, DE, dec_res_b,
                                            f_t, nullptr, DE,
                                            BATCH, DE, DE, nullptr, nullptr);
        k_ln_relu_add<<<BATCH, blk, 0, stream>>>(f_t, f_h, h_bf,
                                                 dec_ln_g, dec_ln_b, DE);
    }
    k_gemm<0><<<gOut, blk, 0, stream>>>(h_bf, nullptr, DE, DE, 0,
                                        w_dout, DE, dec_out_b,
                                        f_t, nullptr, DIN,
                                        BATCH, DIN, DE, nullptr, nullptr);
    k_xhat<<<cdiv((size_t)BATCH * DIN, 256), blk, 0, stream>>>(
        f_t, scale, shift, o_xhat, BATCH * DIN, DIN);
}
